// MultiScaleAttention4D_47931835023549
// MI455X (gfx1250) — compile-verified, matrix operand order checked by reading, not run on hardware
//
#include <hip/hip_runtime.h>
#include <hip/hip_bf16.h>

// ---------------------------------------------------------------------------
// MultiScaleAttention4D for MI455X (gfx1250, wave32, WMMA f32_16x16x32_f16)
// B=512, D_IN=D_MODEL=128, HW=256 (16x16), scales: (4x4, ch 0..63) (2x2, ch 64..127)
// ---------------------------------------------------------------------------

typedef _Float16 half8  __attribute__((ext_vector_type(8)));
typedef _Float16 half16 __attribute__((ext_vector_type(16)));
typedef float    floatx8 __attribute__((ext_vector_type(8)));

__device__ __forceinline__ half16 frag_cat(half8 lo, half8 hi) {
    half16 r;
#pragma unroll
    for (int i = 0; i < 8; ++i) { r[i] = lo[i]; r[i + 8] = hi[i]; }
    return r;
}

// Load an A/B fragment row: 16 halves per lane, k = k0 + hi*8 + [0..7] and
// k0 + 16 + hi*8 + [0..7]   (CDNA5 ISA 16-bit A-matrix 16x32 layout)
__device__ __forceinline__ half16 frag_ld(const _Float16* __restrict__ row, int k0, int hi) {
    half8 a = *(const half8*)(row + k0 + hi * 8);
    half8 b = *(const half8*)(row + k0 + 16 + hi * 8);
    return frag_cat(a, b);
}

__device__ __forceinline__ floatx8 wmma16(half16 a, half16 b, floatx8 c) {
    return __builtin_amdgcn_wmma_f32_16x16x32_f16(false, a, false, b, (short)0, c, false, false);
}

// ---------------------------------------------------------------------------
// Workspace layout (byte offsets, 256B aligned)
// ---------------------------------------------------------------------------
#define WS_WQKV16 0u            // 384*128 f16          =  98304 B
#define WS_BQKV   98304u        // 384 f32              =   1536 B
#define WS_WO16   99840u        // 9*128*128 f16        = 294912 B
#define WS_QKV    394752u       // 512*384*256 f16      = 100663296 B
#define WS_Y      101058048u    // 512*256*128 f16      = 33554432 B

// ---------------------------------------------------------------------------
// Kernel 1: weight prep (f32 -> f16), fuse Wq/Wk/Wv, tap-major Wo
// ---------------------------------------------------------------------------
__global__ void msa_prep_weights(const float* __restrict__ Wq, const float* __restrict__ bq,
                                 const float* __restrict__ Wk, const float* __restrict__ bk,
                                 const float* __restrict__ Wv, const float* __restrict__ bv,
                                 const float* __restrict__ Wo,
                                 _Float16* __restrict__ wqkv16, float* __restrict__ bqkv,
                                 _Float16* __restrict__ wo16) {
    int idx = blockIdx.x * 256 + threadIdx.x;
    if (idx < 384 * 128) {
        int row = idx >> 7, c = idx & 127;
        const float* src = (row < 128) ? Wq : (row < 256) ? Wk : Wv;
        wqkv16[idx] = (_Float16)src[(row & 127) * 128 + c];
    }
    if (idx < 384) {
        bqkv[idx] = (idx < 128) ? bq[idx] : (idx < 256) ? bk[idx - 128] : bv[idx - 256];
    }
    if (idx < 9 * 128 * 128) {  // wo16[tap][o][c] <- Wo[o][c][kh][kw]
        int tap = idx / 16384, rem = idx & 16383;
        int o = rem >> 7, c = rem & 127;
        int kh = tap / 3, kw = tap % 3;
        wo16[idx] = (_Float16)Wo[((o * 128 + c) * 3 + kh) * 3 + kw];
    }
}

// ---------------------------------------------------------------------------
// Kernel 2: QKV projection.  Per batch: QKV(384x256) = Wqkv(384x128) @ X(128x256)
// LDS: X staged as (pixel, channel) f16, row stride 136 halves.
// ---------------------------------------------------------------------------
__global__ void msa_qkv(const float* __restrict__ x, const _Float16* __restrict__ wqkv16,
                        const float* __restrict__ bqkv, _Float16* __restrict__ qkv) {
    extern __shared__ __align__(16) char smem[];
    _Float16* xs = (_Float16*)smem;             // 256 x 136 halves = 69632 B
    const int tid = threadIdx.x;
    const int b = blockIdx.x;

    // stage + convert: x[b][c][p] -> xs[p][c]
    for (int idx = tid; idx < 128 * 256; idx += 256) {
        int c = idx >> 8, p = idx & 255;        // consecutive p -> coalesced global
        xs[p * 136 + c] = (_Float16)x[((size_t)b * 128 + c) * 256 + p];
    }
    __syncthreads();

    const int wave = tid >> 5, lane = tid & 31;
    const int m16 = lane & 15, hi = lane >> 4;

    for (int mt = wave; mt < 24; mt += 8) {     // 24 M-tiles over 8 waves
        const int m0 = mt * 16;
        const _Float16* arow = wqkv16 + (m0 + m16) * 128;
        half16 A[4];
#pragma unroll
        for (int kc = 0; kc < 4; ++kc) A[kc] = frag_ld(arow, kc * 32, hi);

        for (int nt = 0; nt < 16; ++nt) {
            const int n0 = nt * 16;
            const _Float16* brow = xs + (n0 + m16) * 136;
            floatx8 C = {};
#pragma unroll
            for (int kc = 0; kc < 4; ++kc)
                C = wmma16(A[kc], frag_ld(brow, kc * 32, hi), C);
#pragma unroll
            for (int r = 0; r < 8; ++r) {
                int m = m0 + hi * 8 + r;
                qkv[((size_t)b * 384 + m) * 256 + n0 + m16] = (_Float16)(C[r] + bqkv[m]);
            }
        }
    }
}

// ---------------------------------------------------------------------------
// Kernel 3: attention, scale 0 (4x4 patches, ch 0..63): 16 tokens, dim 1024
// scores(16x16) = qt @ kt^T * 32 ; softmax ; y = attn @ vt
// ---------------------------------------------------------------------------
__global__ void msa_attn_s0(const _Float16* __restrict__ qkv, _Float16* __restrict__ Yb) {
    extern __shared__ __align__(16) char smem[];
    _Float16* qs    = (_Float16*)smem;                         // 16 x 1032 h = 33024 B
    _Float16* ks    = (_Float16*)(smem + 33024);               // 33024 B
    _Float16* vT    = (_Float16*)(smem + 66048);               // 1024 x 24 h = 49152 B
    float*    part  = (float*)(smem + 115200);                 // 8 x 256 f32 = 8192 B
    float*    S     = (float*)(smem + 123392);                 // 256 f32 = 1024 B
    _Float16* attnH = (_Float16*)(smem + 124416);              // 16 x 32 h = 1024 B

    const int tid = threadIdx.x, b = blockIdx.x;
    const _Float16* base = qkv + (size_t)b * 384 * 256;

    // stage tokens: d = c*16 + i*4 + j ; t = oh*4 + ow ; pixel = (4oh+i)*16 + 4ow+j
    for (int idx = tid; idx < 16 * 1024; idx += 256) {
        int t = idx >> 10, d = idx & 1023;
        int c = d >> 4, rem = d & 15, i = rem >> 2, j = rem & 3;
        int oh = t >> 2, ow = t & 3;
        int p = (oh * 4 + i) * 16 + (ow * 4 + j);
        qs[t * 1032 + d] = base[c * 256 + p];
        ks[t * 1032 + d] = base[(128 + c) * 256 + p];
        vT[d * 24 + t]   = base[(256 + c) * 256 + p];
    }
    __syncthreads();

    const int wave = tid >> 5, lane = tid & 31;
    const int m16 = lane & 15, hi = lane >> 4;

    // scores: split K=1024 over 8 waves (4 chunks each), reduce via LDS
    {
        floatx8 C = {};
#pragma unroll
        for (int kk = 0; kk < 4; ++kk) {
            int k0 = (wave * 4 + kk) * 32;
            half16 A = frag_ld(qs + m16 * 1032, k0, hi);
            half16 B = frag_ld(ks + m16 * 1032, k0, hi);
            C = wmma16(A, B, C);
        }
#pragma unroll
        for (int r = 0; r < 8; ++r) part[wave * 256 + (hi * 8 + r) * 16 + m16] = C[r];
    }
    __syncthreads();
    {   // reduce partials, apply faithful scale * sqrt(1024) = 32
        float s = 0.f;
#pragma unroll
        for (int w = 0; w < 8; ++w) s += part[w * 256 + tid];
        S[tid] = s * 32.0f;
    }
    __syncthreads();
    if (tid < 16) {  // softmax over each 16-wide row; pad attn to K=32 with zeros
        float mx = -3.402823e38f;
        for (int n = 0; n < 16; ++n) mx = fmaxf(mx, S[tid * 16 + n]);
        float sum = 0.f; float e[16];
        for (int n = 0; n < 16; ++n) { e[n] = __expf(S[tid * 16 + n] - mx); sum += e[n]; }
        float inv = 1.0f / sum;
        for (int n = 0; n < 16; ++n) attnH[tid * 32 + n] = (_Float16)(e[n] * inv);
        for (int n = 16; n < 32; ++n) attnH[tid * 32 + n] = (_Float16)0.f;
    }
    __syncthreads();

    // y(16 x 1024) = attn(16x16, padded K=32) @ vt ; 64 n-tiles over 8 waves
    {
        half16 A = frag_ld(attnH + m16 * 32, 0, hi);
        half8 z = {};
        for (int nn = 0; nn < 8; ++nn) {
            int n0 = (wave * 8 + nn) * 16;
            int d = n0 + m16;
            half8 blo = *(const half8*)(vT + d * 24 + hi * 8);  // k(token) contiguous
            half16 B = frag_cat(blo, z);                        // k>=16 is zero-padded
            floatx8 C = {};
            C = wmma16(A, B, C);
#pragma unroll
            for (int r = 0; r < 8; ++r) {
                int t = hi * 8 + r;
                int dd = n0 + m16;
                int c = dd >> 4, rem = dd & 15, i = rem >> 2, j = rem & 3;
                int oh = t >> 2, ow = t & 3;
                int p = (oh * 4 + i) * 16 + (ow * 4 + j);
                Yb[((size_t)b * 256 + p) * 128 + c] = (_Float16)C[r];
            }
        }
    }
}

// ---------------------------------------------------------------------------
// Kernel 4: attention, scale 1 (2x2 patches, ch 64..127): 64 tokens, dim 256
// ---------------------------------------------------------------------------
__global__ void msa_attn_s1(const _Float16* __restrict__ qkv, _Float16* __restrict__ Yb) {
    extern __shared__ __align__(16) char smem[];
    _Float16* qs    = (_Float16*)smem;                         // 64 x 264 h = 33792 B
    _Float16* ks    = (_Float16*)(smem + 33792);               // 33792 B
    _Float16* vT    = (_Float16*)(smem + 67584);               // 256 x 72 h = 36864 B
    float*    S     = (float*)(smem + 104448);                 // 64 x 64 f32 = 16384 B
    _Float16* attnH = (_Float16*)(smem + 120832);              // 64 x 72 h = 9216 B

    const int tid = threadIdx.x, b = blockIdx.x;
    const _Float16* base = qkv + (size_t)b * 384 * 256;

    // d = c*4 + i*2 + j ; t = oh*8 + ow ; pixel = (2oh+i)*16 + 2ow+j
    for (int idx = tid; idx < 64 * 256; idx += 256) {
        int t = idx >> 8, d = idx & 255;
        int c = d >> 2, rem = d & 3, i = rem >> 1, j = rem & 1;
        int oh = t >> 3, ow = t & 7;
        int p = (oh * 2 + i) * 16 + (ow * 2 + j);
        qs[t * 264 + d] = base[(64 + c) * 256 + p];
        ks[t * 264 + d] = base[(192 + c) * 256 + p];
        vT[d * 72 + t]  = base[(320 + c) * 256 + p];
    }
    __syncthreads();

    const int wave = tid >> 5, lane = tid & 31;
    const int m16 = lane & 15, hi = lane >> 4;

    // scores 64x64: 16 tiles over 8 waves (2 each), full K=256 per tile
    for (int tt = wave * 2; tt < wave * 2 + 2; ++tt) {
        int mt = tt >> 2, nt = tt & 3;
        floatx8 C = {};
#pragma unroll
        for (int kc = 0; kc < 8; ++kc) {
            half16 A = frag_ld(qs + (mt * 16 + m16) * 264, kc * 32, hi);
            half16 B = frag_ld(ks + (nt * 16 + m16) * 264, kc * 32, hi);
            C = wmma16(A, B, C);
        }
#pragma unroll
        for (int r = 0; r < 8; ++r)
            S[(mt * 16 + hi * 8 + r) * 64 + nt * 16 + m16] = C[r] * 16.0f;  // * sqrt(256)
    }
    __syncthreads();
    if (tid < 64) {  // softmax over 64-wide rows -> f16 attn
        float mx = -3.402823e38f;
        for (int n = 0; n < 64; ++n) mx = fmaxf(mx, S[tid * 64 + n]);
        float sum = 0.f;
        for (int n = 0; n < 64; ++n) { float e = __expf(S[tid * 64 + n] - mx); S[tid * 64 + n] = e; sum += e; }
        float inv = 1.0f / sum;
        for (int n = 0; n < 64; ++n) attnH[tid * 72 + n] = (_Float16)(S[tid * 64 + n] * inv);
    }
    __syncthreads();

    // y(64 x 256) = attn(64x64) @ vt(64x256): 4x16 = 64 tiles over 8 waves, K=64
    for (int tt = wave * 8; tt < wave * 8 + 8; ++tt) {
        int mt = tt >> 4, nt = tt & 15;
        floatx8 C = {};
#pragma unroll
        for (int kc = 0; kc < 2; ++kc) {
            half16 A = frag_ld(attnH + (mt * 16 + m16) * 72, kc * 32, hi);
            half16 B = frag_ld(vT + (nt * 16 + m16) * 72, kc * 32, hi);
            C = wmma16(A, B, C);
        }
#pragma unroll
        for (int r = 0; r < 8; ++r) {
            int t = mt * 16 + hi * 8 + r;
            int d = nt * 16 + m16;
            int c = d >> 2, rem = d & 3, i = rem >> 1, j = rem & 1;
            int oh = t >> 3, ow = t & 7;
            int p = (oh * 2 + i) * 16 + (ow * 2 + j);
            Yb[((size_t)b * 256 + p) * 128 + 64 + c] = (_Float16)C[r];
        }
    }
}

// ---------------------------------------------------------------------------
// Kernel 5: 3x3 conv + bias + LeakyReLU(0.2) as implicit GEMM over 9 taps
// LDS: zero-padded 18x18 image, (padded pixel, channel), stride 136 halves
// ---------------------------------------------------------------------------
__global__ void msa_conv3x3(const _Float16* __restrict__ Yb, const _Float16* __restrict__ wo16,
                            const float* __restrict__ bo, float* __restrict__ out) {
    extern __shared__ __align__(16) char smem[];
    _Float16* ys = (_Float16*)smem;             // 324 x 136 halves = 88128 B
    const int tid = threadIdx.x, b = blockIdx.x;

    for (int idx = tid; idx < 324 * 128; idx += 256) {
        int pp = idx >> 7, c = idx & 127;
        int hh = pp / 18, ww = pp % 18;
        _Float16 v = (_Float16)0.f;
        if (hh >= 1 && hh <= 16 && ww >= 1 && ww <= 16)
            v = Yb[((size_t)b * 256 + (hh - 1) * 16 + (ww - 1)) * 128 + c];
        ys[pp * 136 + c] = v;
    }
    __syncthreads();

    const int wave = tid >> 5, lane = tid & 31;
    const int m16 = lane & 15, hi = lane >> 4;
    const int o0 = wave * 16;                   // one M-tile of out-channels per wave

    for (int nt = 0; nt < 16; ++nt) {
        const int p = nt * 16 + m16;
        const int h = p >> 4, w = p & 15;
        floatx8 C = {};
#pragma unroll
        for (int tap = 0; tap < 9; ++tap) {
            const int dh = tap / 3, dw = tap % 3;
            const _Float16* brow = ys + ((h + dh) * 18 + (w + dw)) * 136;
            const _Float16* arow = wo16 + ((size_t)tap * 128 + o0 + m16) * 128;
#pragma unroll
            for (int kc = 0; kc < 4; ++kc)
                C = wmma16(frag_ld(arow, kc * 32, hi), frag_ld(brow, kc * 32, hi), C);
        }
#pragma unroll
        for (int r = 0; r < 8; ++r) {
            int o = o0 + hi * 8 + r;
            float v = C[r] + bo[o];
            v = (v >= 0.f) ? v : 0.2f * v;
            out[((size_t)b * 128 + o) * 256 + p] = v;
        }
    }
}

// ---------------------------------------------------------------------------
// Host launcher
// ---------------------------------------------------------------------------
extern "C" void kernel_launch(void* const* d_in, const int* in_sizes, int n_in,
                              void* d_out, int out_size, void* d_ws, size_t ws_size,
                              hipStream_t stream) {
    const float* x  = (const float*)d_in[0];
    const float* Wq = (const float*)d_in[1];
    const float* bq = (const float*)d_in[2];
    const float* Wk = (const float*)d_in[3];
    const float* bk = (const float*)d_in[4];
    const float* Wv = (const float*)d_in[5];
    const float* bv = (const float*)d_in[6];
    const float* Wo = (const float*)d_in[7];
    const float* bo = (const float*)d_in[8];
    float* out = (float*)d_out;

    char* ws = (char*)d_ws;
    _Float16* wqkv16 = (_Float16*)(ws + WS_WQKV16);
    float*    bqkv   = (float*)(ws + WS_BQKV);
    _Float16* wo16   = (_Float16*)(ws + WS_WO16);
    _Float16* qkv    = (_Float16*)(ws + WS_QKV);
    _Float16* Yb     = (_Float16*)(ws + WS_Y);

    // allow >64KB dynamic LDS (WGP has 320KB); deterministic, capture-safe
    (void)hipFuncSetAttribute((const void*)msa_qkv,     hipFuncAttributeMaxDynamicSharedMemorySize, 69632);
    (void)hipFuncSetAttribute((const void*)msa_attn_s0, hipFuncAttributeMaxDynamicSharedMemorySize, 125440);
    (void)hipFuncSetAttribute((const void*)msa_attn_s1, hipFuncAttributeMaxDynamicSharedMemorySize, 130048);
    (void)hipFuncSetAttribute((const void*)msa_conv3x3, hipFuncAttributeMaxDynamicSharedMemorySize, 88128);

    msa_prep_weights<<<576, 256, 0, stream>>>(Wq, bq, Wk, bk, Wv, bv, Wo, wqkv16, bqkv, wo16);
    msa_qkv<<<512, 256, 69632, stream>>>(x, wqkv16, bqkv, qkv);
    msa_attn_s0<<<512, 256, 125440, stream>>>(qkv, Yb);
    msa_attn_s1<<<512, 256, 130048, stream>>>(qkv, Yb);
    msa_conv3x3<<<512, 256, 88128, stream>>>(Yb, wo16, bo, out);
}
